// ArgmaxISAModule_23347442221062
// MI455X (gfx1250) — compile-verified, matrix operand order checked by reading, not run on hardware
//
#include <hip/hip_runtime.h>

// ArgmaxISAModule for MI455X (gfx1250): bf16 WMMA GEMM pipeline, fp32 accumulate.
// L=8 layers, H=4 heads, Q=64, D=512, N=2048, DFF=2048.
// 128x128 block tile, 256 threads = 8 wave32s, each wave owns 64x32 (4x2 fragments).

typedef __attribute__((ext_vector_type(16))) __bf16 v16bf;
typedef __attribute__((ext_vector_type(8)))  __bf16 v8bf;
typedef __attribute__((ext_vector_type(8)))  float  v8f;
typedef __attribute__((ext_vector_type(4)))  unsigned short us4;

#define BM 128
#define BN 128
#define BK 32
#define TSTRIDE 40  // padded LDS row stride in bf16 elems (20 dwords -> conflict-free frag reads)

__device__ __forceinline__ unsigned short f2bf(float f) {
  unsigned u = __float_as_uint(f);
  u += 0x7FFFu + ((u >> 16) & 1u);   // round-to-nearest-even to bf16
  return (unsigned short)(u >> 16);
}

// Load one 16x32 bf16 fragment (A-style; B uses same layout on an [n][k] tile).
// ISA layout: lanes 0-15: row=lane, K0..7 -> v0..3, K16..23 -> v4..7;
//             lanes 16-31: row=lane-16, K8..15 -> v0..3, K24..31 -> v4..7.
__device__ __forceinline__ v16bf load_frag(const unsigned short* tile, int rowBase, int lane) {
  const int half = lane >> 4;
  const int row  = rowBase + (lane & 15);
  const unsigned short* p = tile + row * TSTRIDE;
  v8bf lo = *(const v8bf*)(const void*)(p + half * 8);
  v8bf hi = *(const v8bf*)(const void*)(p + 16 + half * 8);
  return __builtin_shufflevector(lo, hi, 0,1,2,3,4,5,6,7,8,9,10,11,12,13,14,15);
}

// Generic C = A*B GEMM with optional per-row bias, elementwise add-matrix, relu.
// A: MxK (row-major, lda) or KxM if transA (then element (m,k) = A[k*lda+m]).
// B: KxN row-major (ldb), N must be a multiple of BN. K multiple of BK.
// M may be ragged (clamped+zero-selected staging, guarded epilogue).
// All base pointers / leading dims must be 16B-aligned (true for this pipeline).
// Batch via blockIdx.z with strides sA/sB/sC.
__global__ __launch_bounds__(256)
void gemm_bf16_wmma(const float* __restrict__ A, const float* __restrict__ B,
                    float* __restrict__ C,
                    const float* __restrict__ bias, const float* __restrict__ add,
                    int M, int N, int Kd, int lda, int ldb, int ldc, int ldadd,
                    long sA, long sB, long sC, int transA, int relu)
{
  __shared__ __align__(16) unsigned short At[BM * TSTRIDE];
  __shared__ __align__(16) unsigned short Bt[BN * TSTRIDE];

  A += (long)blockIdx.z * sA;
  B += (long)blockIdx.z * sB;
  C += (long)blockIdx.z * sC;

  const int tid   = threadIdx.x;
  const int wave  = tid >> 5;           // wave32
  const int lane  = tid & 31;
  const int wm    = (wave >> 2) * 64;   // 2 wave-rows x 4 wave-cols over 128x128
  const int wn    = (wave & 3) * 32;
  const int tileM = blockIdx.y * BM;
  const int tileN = blockIdx.x * BN;

  v8f acc[4][2] = {};

  for (int k0 = 0; k0 < Kd; k0 += BK) {
    // --- stage A tile (fp32 -> bf16), branchless ragged-M handling ---
    if (!transA) {
      // float4 over k (coalesced, 16B), packed ds_store_b64
      for (int g = tid; g < BM * (BK / 4); g += 256) {    // 1024 groups
        const int m  = g >> 3;
        const int kq = (g & 7) * 4;
        const int gm = tileM + m;
        const int rc = (gm < M) ? gm : (M - 1);           // clamp: always in-bounds
        const float4 v = *(const float4*)(A + (long)rc * lda + (k0 + kq));
        us4 h;
        if (gm < M) {
          h.x = f2bf(v.x); h.y = f2bf(v.y); h.z = f2bf(v.z); h.w = f2bf(v.w);
        } else {
          h.x = 0; h.y = 0; h.z = 0; h.w = 0;
        }
        *(us4*)(void*)(&At[m * TSTRIDE + kq]) = h;        // 8B aligned (80B rows, kq%4==0)
      }
    } else {
      // float4 over m (coalesced), 4 scattered b16 stores
      for (int g = tid; g < (BM / 4) * BK; g += 256) {    // 1024 groups
        const int m4 = (g & 31) * 4;
        const int k  = g >> 5;
        const int gm = tileM + m4;
        if (gm + 3 < M) {
          const float4 v = *(const float4*)(A + (long)(k0 + k) * lda + gm);
          At[(m4 + 0) * TSTRIDE + k] = f2bf(v.x);
          At[(m4 + 1) * TSTRIDE + k] = f2bf(v.y);
          At[(m4 + 2) * TSTRIDE + k] = f2bf(v.z);
          At[(m4 + 3) * TSTRIDE + k] = f2bf(v.w);
        } else {
          for (int j = 0; j < 4; ++j) {
            const int gmj = gm + j;
            const int rc  = (gmj < M) ? gmj : (M - 1);
            const float x = A[(long)(k0 + k) * lda + rc];
            At[(m4 + j) * TSTRIDE + k] = (gmj < M) ? f2bf(x) : (unsigned short)0;
          }
        }
      }
    }
    // --- stage B tile, stored [n][k]: float4 over n (coalesced), scattered b16 stores ---
    for (int g = tid; g < (BN / 4) * BK; g += 256) {      // 1024 groups
      const int n4 = (g & 31) * 4;
      const int k  = g >> 5;
      const float4 v = *(const float4*)(B + (long)(k0 + k) * ldb + (tileN + n4));
      Bt[(n4 + 0) * TSTRIDE + k] = f2bf(v.x);
      Bt[(n4 + 1) * TSTRIDE + k] = f2bf(v.y);
      Bt[(n4 + 2) * TSTRIDE + k] = f2bf(v.z);
      Bt[(n4 + 3) * TSTRIDE + k] = f2bf(v.w);
    }
    __syncthreads();

    // Prefetch next K-tile into cache while WMMAs run (global_prefetch_b8).
    if (k0 + BK < Kd) {
      const int kn = k0 + BK;
      if (!transA) {
        const int m = tid >> 1, seg = tid & 1;
        const int rc = (tileM + m < M) ? (tileM + m) : (M - 1);
        __builtin_prefetch(A + (long)rc * lda + kn + seg * 16, 0, 1);
      } else if (tid < 128) {
        const int k = tid >> 2, seg = tid & 3;
        __builtin_prefetch(A + (long)(kn + k) * lda + tileM + seg * 32, 0, 1);
      }
      if (tid < 128) {
        const int k = tid >> 2, seg = tid & 3;
        __builtin_prefetch(B + (long)(kn + k) * ldb + tileN + seg * 32, 0, 1);
      }
    }

    v16bf b0 = load_frag(Bt, wn,      lane);
    v16bf b1 = load_frag(Bt, wn + 16, lane);
#pragma unroll
    for (int i = 0; i < 4; ++i) {
      v16bf a = load_frag(At, wm + i * 16, lane);
      acc[i][0] = __builtin_amdgcn_wmma_f32_16x16x32_bf16(false, a, false, b0, (short)0, acc[i][0], false, false);
      acc[i][1] = __builtin_amdgcn_wmma_f32_16x16x32_bf16(false, a, false, b1, (short)0, acc[i][1], false, false);
    }

    __syncthreads();
  }

  // Epilogue. C/D layout: lane 0-15 -> n=lane, m=r; lane 16-31 -> n=lane-16, m=8+r.
  const int nLoc = lane & 15;
  const int mHi  = (lane >> 4) * 8;
#pragma unroll
  for (int i = 0; i < 4; ++i)
#pragma unroll
    for (int j = 0; j < 2; ++j) {
      const int gn = tileN + wn + j * 16 + nLoc;
#pragma unroll
      for (int r = 0; r < 8; ++r) {
        const int gm = tileM + wm + i * 16 + mHi + r;
        if (gm < M) {
          float v = acc[i][j][r];
          if (bias) v += bias[gm];
          if (add)  v += add[(long)gm * ldadd + gn];
          if (relu) v = v > 0.f ? v : 0.f;
          C[(long)gm * ldc + gn] = v;
        }
      }
    }
}

__global__ __launch_bounds__(256)
void rowmax_kernel(const float* __restrict__ S, float* __restrict__ rmax, int N) {
  __shared__ float red[256];
  const int row = blockIdx.x;
  const float* p = S + (long)row * N;
  float m = -3.402823466e38f;
  for (int i = threadIdx.x; i < N; i += 256) m = fmaxf(m, p[i]);
  red[threadIdx.x] = m; __syncthreads();
  for (int s = 128; s > 0; s >>= 1) {
    if (threadIdx.x < s) red[threadIdx.x] = fmaxf(red[threadIdx.x], red[threadIdx.x + s]);
    __syncthreads();
  }
  if (threadIdx.x == 0) rmax[row] = red[0];
}

// In-place: scores row -> weights row (mask / rowsum * active)
__global__ __launch_bounds__(256)
void weights_kernel(float* __restrict__ S, const float* __restrict__ rmax, int N) {
  __shared__ float red[256];
  const int row = blockIdx.x;
  const float mv  = rmax[row];
  const float thr = mv - 0.5f;
  float* p = S + (long)row * N;
  float cnt = 0.f;
  for (int i = threadIdx.x; i < N; i += 256) cnt += (p[i] >= thr) ? 1.f : 0.f;
  red[threadIdx.x] = cnt; __syncthreads();
  for (int s = 128; s > 0; s >>= 1) {
    if (threadIdx.x < s) red[threadIdx.x] += red[threadIdx.x + s];
    __syncthreads();
  }
  const float total  = red[0];
  const float active = (fabsf(mv) > 0.5f) ? 1.f : 0.f;
  const float w = active / fmaxf(total, 1.f);
  for (int i = threadIdx.x; i < N; i += 256)
    p[i] = (p[i] >= thr) ? w : 0.f;
}

extern "C" void kernel_launch(void* const* d_in, const int* in_sizes, int n_in,
                              void* d_out, int out_size, void* d_ws, size_t ws_size,
                              hipStream_t stream) {
  const int L = 8, H = 4, Q = 64, D = 512, Nn = 2048, DFF = 2048;
  const float* Xin = (const float*)d_in[0];
  const float* Kw  = (const float*)d_in[1];
  const float* Vw  = (const float*)d_in[2];
  const float* W1  = (const float*)d_in[3];
  const float* B1  = (const float*)d_in[4];
  const float* W2  = (const float*)d_in[5];
  const float* B2  = (const float*)d_in[6];

  // Workspace layout (fp32), ~48 MB total.
  float* ws = (float*)d_ws;
  size_t o = 0;
  float* Xcur = ws + o; o += (size_t)D * Nn;       // current residual stream
  float* KX   = ws + o; o += (size_t)H * Q * Nn;   // all heads' K@X
  float* S    = ws + o; o += (size_t)Nn * Nn;      // scores -> weights (per head, reused)
  float* Xw   = ws + o; o += (size_t)D * Nn;       // X @ weights (per head)
  float* attn = ws + o; o += (size_t)D * Nn;       // residual + sum_h V@Xw
  float* ff1  = ws + o; o += (size_t)DFF * Nn;     // relu(W1@attn + b1)
  float* rmax = ws + o; o += (size_t)Nn;

  hipMemcpyAsync(Xcur, Xin, sizeof(float) * (size_t)D * Nn, hipMemcpyDeviceToDevice, stream);

  const dim3 blk(256);
  auto gridFor = [](int M, int N, int z) {
    return dim3((unsigned)(N / BN), (unsigned)((M + BM - 1) / BM), (unsigned)z);
  };

  for (int l = 0; l < L; ++l) {
    const float* Kl = Kw + (size_t)l * H * Q * D;

    // KX[h] = K[l,h] (QxD) @ Xcur (DxN), batched over heads via grid.z
    gemm_bf16_wmma<<<gridFor(Q, Nn, H), blk, 0, stream>>>(
        Kl, Xcur, KX, nullptr, nullptr,
        Q, Nn, D, D, Nn, Nn, 0,
        (long)Q * D, 0L, (long)Q * Nn, 0, 0);

    // attn starts as residual copy of Xcur
    hipMemcpyAsync(attn, Xcur, sizeof(float) * (size_t)D * Nn, hipMemcpyDeviceToDevice, stream);

    for (int h = 0; h < H; ++h) {
      const float* KXh = KX + (size_t)h * Q * Nn;

      // scores = KX^T (NxQ, via transA) @ KX (QxN)
      gemm_bf16_wmma<<<gridFor(Nn, Nn, 1), blk, 0, stream>>>(
          KXh, KXh, S, nullptr, nullptr,
          Nn, Nn, Q, Nn, Nn, Nn, 0,
          0L, 0L, 0L, 1, 0);

      rowmax_kernel<<<Nn, 256, 0, stream>>>(S, rmax, Nn);
      weights_kernel<<<Nn, 256, 0, stream>>>(S, rmax, Nn);

      // Xw = Xcur (DxN) @ weights (NxN)
      gemm_bf16_wmma<<<gridFor(D, Nn, 1), blk, 0, stream>>>(
          Xcur, S, Xw, nullptr, nullptr,
          D, Nn, Nn, Nn, Nn, Nn, 0,
          0L, 0L, 0L, 0, 0);

      // attn += V[l,h] (DxD) @ Xw (DxN)   (add = attn -> accumulate)
      const float* Vh = Vw + (size_t)(l * H + h) * D * D;
      gemm_bf16_wmma<<<gridFor(D, Nn, 1), blk, 0, stream>>>(
          Vh, Xw, attn, nullptr, attn,
          D, Nn, D, D, Nn, Nn, Nn,
          0L, 0L, 0L, 0, 0);
    }

    // ff1 = relu(W1[l] @ attn + b1[l])
    gemm_bf16_wmma<<<gridFor(DFF, Nn, 1), blk, 0, stream>>>(
        W1 + (size_t)l * DFF * D, attn, ff1, B1 + (size_t)l * DFF, nullptr,
        DFF, Nn, D, D, Nn, Nn, 0,
        0L, 0L, 0L, 0, 1);

    // Xcur = attn + W2[l] @ ff1 + b2[l]
    gemm_bf16_wmma<<<gridFor(D, Nn, 1), blk, 0, stream>>>(
        W2 + (size_t)l * D * DFF, ff1, Xcur, B2 + (size_t)l * D, attn,
        D, Nn, DFF, DFF, Nn, Nn, Nn,
        0L, 0L, 0L, 0, 0);
  }

  hipMemcpyAsync(d_out, Xcur, sizeof(float) * (size_t)D * Nn, hipMemcpyDeviceToDevice, stream);
}
